// FloatingRegionScore_7335804141612
// MI455X (gfx1250) — 8-bit matrix operand order repaired and checked by reading, compiled, not run on hardware
//
#include <hip/hip_runtime.h>
#include <hip/hip_bf16.h>
#include <math.h>


#define W    2048
#define H    1024
#define HW   (W*H)
#define C19  19

#define WPF  2096
#define WPB  2112
#define PLANE (H*WPB)

typedef __attribute__((ext_vector_type(2))) float v2f;
typedef __attribute__((ext_vector_type(8))) float v8f;
typedef __attribute__((ext_vector_type(8))) int   v8i;

#define LOG2E      1.4426950408889634f
#define LN2_LN19   0.23540891336663824f

__global__ void __launch_bounds__(256)
k_pixel(const float* __restrict__ logit, const float* __restrict__ cw,
        unsigned char* __restrict__ lbl, float* __restrict__ wpe) {
  int i = blockIdx.x * 256 + threadIdx.x;
  float x[C19];
  #pragma unroll
  for (int c = 0; c < C19; ++c)
    x[c] = __builtin_nontemporal_load(logit + c * HW + i);
  float m = x[0]; int am = 0;
  #pragma unroll
  for (int c = 1; c < C19; ++c) if (x[c] > m) { m = x[c]; am = c; }
  float S = 0.f;
  #pragma unroll
  for (int c = 0; c < C19; ++c) {
    x[c] = __builtin_amdgcn_exp2f((x[c] - m) * LOG2E); S += x[c];
  }
  float inv = 1.f / S, ent2 = 0.f;
  #pragma unroll
  for (int c = 0; c < C19; ++c) {
    float p = x[c] * inv;
    ent2 -= p * __builtin_amdgcn_logf(p + 1e-6f);
  }
  lbl[i] = (unsigned char)am;
  wpe[i] = ent2 * LN2_LN19 * cw[am];
}

__global__ void __launch_bounds__(256)
k_vert(const unsigned char* __restrict__ lbl, const float* __restrict__ wpe,
       unsigned char* __restrict__ histVp, float* __restrict__ wpeVp) {
  int t  = blockIdx.x * 256 + threadIdx.x;
  int x  = t & (W - 1);
  int y0 = (t >> 11) * 128;
  int c  = blockIdx.y;
  if (c < C19) {
    int cnt = 0;
    for (int j = y0 - 16; j <= y0 + 16; ++j)
      if (j >= 0 && j < H) cnt += (lbl[j * W + x] == c);
    unsigned char* o = histVp + c * PLANE;
    for (int y = y0; y < y0 + 128; ++y) {
      o[y * WPB + 16 + x] = (unsigned char)cnt;
      int ya = y + 17; if (ya < H) cnt += (lbl[ya * W + x] == c);
      int yr = y - 16; if (yr >= 0) cnt -= (lbl[yr * W + x] == c);
    }
  } else {
    for (int y = y0; y < y0 + 128; ++y) {
      float s = 0.f;
      int jl = max(y - 16, 0), jh = min(y + 16, H - 1);
      for (int j = jl; j <= jh; ++j) s += wpe[j * W + x];
      wpeVp[y * WPF + 16 + x] = s;
    }
  }
}

__global__ void __launch_bounds__(256)
k_final(const unsigned char* __restrict__ histVp, const float* __restrict__ wpeVp,
        float* __restrict__ out) {
  __shared__ float sh[8][16][17];
  int lane = threadIdx.x & 31;
  int w    = threadIdx.x >> 5;
  int tIdx = blockIdx.x * 8 + w;
  int x0   = (tIdx & 127) * 16;
  int y0   = (tIdx >> 7) * 16;
  int n    = lane & 15;
  int g    = lane >> 4;
  int m    = n;

  v8f dW = {};
  const float* wrow = wpeVp + (y0 + n) * WPF + x0;
  #pragma unroll
  for (int q = 0; q < 12; ++q) {
    int j0 = 4 * q + 2 * g;
    v2f a;
    a.x = (j0     >= m && j0     <= m + 32) ? 1.f : 0.f;
    a.y = (j0 + 1 >= m && j0 + 1 <= m + 32) ? 1.f : 0.f;
    v2f b = *(const v2f*)(wrow + j0);
    dW = __builtin_amdgcn_wmma_f32_16x16x4_f32(false, a, false, b,
                                               (short)0, dW, false, false);
  }

  v8i A8;
  #pragma unroll
  for (int v = 0; v < 8; ++v) {
    unsigned int word = 0;
    #pragma unroll
    for (int b = 0; b < 4; ++b) {
      int K = 32 * (v >> 2) + 16 * g + 4 * (v & 3) + b;
      if (K >= m && K <= m + 32) word |= (1u << (8 * b));
    }
    A8[v] = (int)word;
  }

  float cyv = (float)(min(y0 + n + 16, H - 1) - max(y0 + n - 16, 0) + 1);
  float rinv[8];
  #pragma unroll
  for (int r = 0; r < 8; ++r) {
    int xx = x0 + r + 8 * g;
    float cx = (float)(min(xx + 16, W - 1) - max(xx - 16, 0) + 1);
    rinv[r] = 1.f / (cx * cyv);
  }

  float impAcc[8];
  #pragma unroll
  for (int r = 0; r < 8; ++r) impAcc[r] = 0.f;
  const unsigned char* hv = histVp + (y0 + n) * WPB + x0 + 16 * g;
  uint4 lo = *(const uint4*)(hv);
  uint4 hi = *(const uint4*)(hv + 32);

  v8i B8;
  B8[0] = (int)lo.x; B8[1] = (int)lo.y; B8[2] = (int)lo.z; B8[3] = (int)lo.w;
  B8[4] = (int)hi.x; B8[5] = (int)hi.y; B8[6] = (int)hi.z; B8[7] = (int)hi.w;
  v8i zc = {};
  v8i dPrev = __builtin_amdgcn_wmma_i32_16x16x64_iu8(false, A8, false, B8,
                                                     zc, false, false);
  hv += PLANE;
  lo = *(const uint4*)(hv);
  hi = *(const uint4*)(hv + 32);

  for (int c = 1; c < C19; ++c) {
    B8[0] = (int)lo.x; B8[1] = (int)lo.y; B8[2] = (int)lo.z; B8[3] = (int)lo.w;
    B8[4] = (int)hi.x; B8[5] = (int)hi.y; B8[6] = (int)hi.z; B8[7] = (int)hi.w;
    v8i dCur = __builtin_amdgcn_wmma_i32_16x16x64_iu8(false, A8, false, B8,
                                                      zc, false, false);
    hv += PLANE;
    lo = *(const uint4*)(hv);
    hi = *(const uint4*)(hv + 32);
    #pragma unroll
    for (int r = 0; r < 8; ++r) {
      float d = (float)dPrev[r] * rinv[r];
      impAcc[r] -= d * __builtin_amdgcn_logf(d + 1e-6f);
    }
    dPrev = dCur;
  }
  #pragma unroll
  for (int r = 0; r < 8; ++r) {
    float d = (float)dPrev[r] * rinv[r];
    impAcc[r] -= d * __builtin_amdgcn_logf(d + 1e-6f);
  }

  float scv[8], impv[8], uncv[8];
  #pragma unroll
  for (int r = 0; r < 8; ++r) {
    float imp = impAcc[r] * LN2_LN19;
    float unc = dW[r] * rinv[r];
    impv[r] = imp; uncv[r] = unc; scv[r] = imp * unc;
  }
  #pragma unroll
  for (int q3 = 0; q3 < 3; ++q3) {
    __syncthreads();
    #pragma unroll
    for (int r = 0; r < 8; ++r) {
      float v = (q3 == 0) ? scv[r] : (q3 == 1) ? impv[r] : uncv[r];
      sh[w][n][r + 8 * g] = v;
    }
    __syncthreads();
    #pragma unroll
    for (int j = 0; j < 8; ++j) {
      int row = 2 * j + g;
      out[q3 * HW + (y0 + row) * W + x0 + n] = sh[w][row][n];
    }
  }
}

extern "C" void kernel_launch(void* const* d_in, const int* in_sizes, int n_in,
                              void* d_out, int out_size, void* d_ws, size_t ws_size,
                              hipStream_t stream) {
  const float* logit = (const float*)d_in[0];
  const float* cw    = (const float*)d_in[1];
  float* out = (float*)d_out;
  char* ws = (char*)d_ws;

  unsigned char* lbl    = (unsigned char*)(ws);
  float*         wpe    = (float*)(ws + (size_t)2  * 1024 * 1024);
  float*         wpeVp  = (float*)(ws + (size_t)10 * 1024 * 1024);
  unsigned char* histVp = (unsigned char*)(ws + (size_t)20 * 1024 * 1024);

  size_t wpeVpBytes  = (size_t)H * WPF * sizeof(float);
  size_t histVpBytes = (size_t)C19 * PLANE;

  hipMemsetAsync(wpeVp,  0, wpeVpBytes,  stream);
  hipMemsetAsync(histVp, 0, histVpBytes, stream);

  k_pixel<<<HW / 256, 256, 0, stream>>>(logit, cw, lbl, wpe);
  k_vert <<<dim3(64, 20), 256, 0, stream>>>(lbl, wpe, histVp, wpeVp);
  k_final<<<(HW / 256) / 8, 256, 0, stream>>>(histVp, wpeVp, out);
}
